// UpsampleTransformer_67903432950524
// MI455X (gfx1250) — matrix operand order repaired and checked by reading, not compiled, not run on hardware
//
#include <hip/hip_runtime.h>
#include <hip/hip_bf16.h>

// ---------------------------------------------------------------------------
// UpsampleTransformer for MI455X (gfx1250, wave32, WMMA bf16 16x16x32)
// ---------------------------------------------------------------------------
#define BB      8
#define NN      2048
#define IND     128
#define DD      64
#define KNN     20
#define UPF     2
#define HID     256
#define RROWS   (BB * NN * KNN)   // 327,680 neighbor rows
#define MROWS   (BB * NN)         // 16,384 point rows
#define EPSV    1e-5f

typedef __attribute__((ext_vector_type(16))) __bf16 v16bf;
typedef __attribute__((ext_vector_type(8)))  __bf16 bf16x8;
typedef __attribute__((ext_vector_type(8)))  float  v8f;

__device__ __forceinline__ v8f wmma_bf16(const v16bf& a, const v16bf& b, const v8f& c) {
  return __builtin_amdgcn_wmma_f32_16x16x32_bf16(false, a, false, b, (short)0, c, false, false);
}

// A fragment: 16x32 bf16 tile, row-major, leading dim `ld` (multiple of 8).
// lane L: row r=L&15, half h=L>>4. Elements 0..15 = row[k=h*16 .. h*16+15]
// -> two 16B contiguous LDS loads.
__device__ __forceinline__ v16bf load_a_frag(const __bf16* base, int ld, int lane) {
  int r = lane & 15, h = lane >> 4;
  const __bf16* p = base + r * ld + h * 16;
  bf16x8 lo = *(const bf16x8*)(p);
  bf16x8 hi = *(const bf16x8*)(p + 8);
  v16bf a;
#pragma unroll
  for (int i = 0; i < 8; ++i) { a[i] = lo[i]; a[8 + i] = hi[i]; }
  return a;
}

// B fragment from fragment-linear packed weights: one 32B load per lane.
// Packed layout: Wp[((tile)*32 + lane)*16 + e], tile = nt*KS + ks,
// element e of lane L = W[ks*32 + (L>>4)*16 + e][nt*16 + (L&15)].
__device__ __forceinline__ v16bf load_b_packed(const __bf16* Wp, int tile, int lane) {
  return *(const v16bf*)(Wp + ((long)tile * 32 + lane) * 16);
}

// ---------------------------------------------------------------------------
// Pack W[K,N] (f32, row-major) -> fragment-linear bf16.
// ---------------------------------------------------------------------------
__global__ void pack_weight(const float* __restrict__ W, __bf16* __restrict__ Wp,
                            int Kd, int Nd) {
  int total = Kd * Nd;
  int KS = Kd >> 5;
  for (int p = blockIdx.x * blockDim.x + threadIdx.x; p < total;
       p += gridDim.x * blockDim.x) {
    int e = p & 15;
    int lane = (p >> 4) & 31;
    int t2 = p >> 9;
    int ks = t2 % KS;
    int nt = t2 / KS;
    int k = ks * 32 + (lane >> 4) * 16 + e;
    int c = nt * 16 + (lane & 15);
    Wp[p] = (__bf16)W[(long)k * Nd + c];
  }
}

// ---------------------------------------------------------------------------
// concat kq = [key_feat | query]  -> [MROWS, 256]
// ---------------------------------------------------------------------------
__global__ void concat_kernel(const float* __restrict__ kf, const float* __restrict__ q,
                              float* __restrict__ kq) {
  for (int i = blockIdx.x * blockDim.x + threadIdx.x; i < MROWS * 256;
       i += gridDim.x * blockDim.x) {
    int row = i >> 8, c = i & 255;
    kq[i] = (c < IND) ? kf[row * IND + c] : q[row * IND + (c - IND)];
  }
}

// ---------------------------------------------------------------------------
// Generic bf16 WMMA GEMM with packed weights:
// C[M,N] = act(A[M,K] @ W[K,N] + bias) (+= C if ACC)
// grid (M/64, N/64), block 128 (4 waves); K multiple of 32.
// ---------------------------------------------------------------------------
template <bool RELU, bool ACC>
__global__ void wmma_gemm(const float* __restrict__ A, const __bf16* __restrict__ Wp,
                          const float* __restrict__ bias, float* __restrict__ C,
                          int M, int Kd, int Nd) {
  __shared__ __bf16 As[64][40];
  int m0 = blockIdx.x * 64;
  int n0t = blockIdx.y * 4;     // base n-tile
  int KS = Kd >> 5;
  int w = threadIdx.x >> 5, lane = threadIdx.x & 31;
  v8f acc[4] = {};
  for (int ks = 0; ks < KS; ++ks) {
    int k0 = ks * 32;
    for (int i = threadIdx.x; i < 64 * 32; i += 128) {
      int r = i >> 5, c = i & 31;
      As[r][c] = (__bf16)A[(long)(m0 + r) * Kd + k0 + c];
    }
    if (k0 + 32 < Kd)  // prefetch next A panel (global_prefetch_b8)
      __builtin_prefetch(&A[(long)(m0 + (threadIdx.x >> 1)) * Kd + k0 + 32], 0, 1);
    __syncthreads();
    v16bf af = load_a_frag(&As[w * 16][0], 40, lane);
#pragma unroll
    for (int nt = 0; nt < 4; ++nt) {
      v16bf bf_ = load_b_packed(Wp, (n0t + nt) * KS + ks, lane);
      acc[nt] = wmma_bf16(af, bf_, acc[nt]);
    }
    __syncthreads();
  }
  int cc = lane & 15, hh = lane >> 4;
#pragma unroll
  for (int nt = 0; nt < 4; ++nt)
#pragma unroll
    for (int v = 0; v < 8; ++v) {
      int row = m0 + w * 16 + v + 8 * hh;
      int col = (n0t + nt) * 16 + cc;
      float x = acc[nt][v] + bias[col];
      if (RELU) x = fmaxf(x, 0.f);
      long o = (long)row * Nd + col;
      if (ACC) x += C[o];
      C[o] = x;
    }
}

// ---------------------------------------------------------------------------
// KNN: per query point, top-20 nearest (self included), register insertion.
// ---------------------------------------------------------------------------
__global__ void knn_kernel(const float* __restrict__ pos, int* __restrict__ idx) {
  __shared__ float sp[128][3];
  int b = blockIdx.x / (NN / 128);
  int n = (blockIdx.x % (NN / 128)) * 128 + threadIdx.x;
  float px = pos[((long)b * NN + n) * 3 + 0];
  float py = pos[((long)b * NN + n) * 3 + 1];
  float pz = pos[((long)b * NN + n) * 3 + 2];
  float best[KNN];
  int bidx[KNN];
#pragma unroll
  for (int i = 0; i < KNN; ++i) { best[i] = 1e30f; bidx[i] = 0; }
  for (int tile = 0; tile < NN; tile += 128) {
    int j = tile + threadIdx.x;
    sp[threadIdx.x][0] = pos[((long)b * NN + j) * 3 + 0];
    sp[threadIdx.x][1] = pos[((long)b * NN + j) * 3 + 1];
    sp[threadIdx.x][2] = pos[((long)b * NN + j) * 3 + 2];
    __syncthreads();
    for (int t = 0; t < 128; ++t) {
      float dx = px - sp[t][0], dy = py - sp[t][1], dz = pz - sp[t][2];
      float d = dx * dx + dy * dy + dz * dz;
      if (d < best[KNN - 1]) {
        float cd = d; int ci = tile + t;
#pragma unroll
        for (int qn = 0; qn < KNN; ++qn) {
          if (cd < best[qn]) {
            float tf = best[qn]; int ti = bidx[qn];
            best[qn] = cd; bidx[qn] = ci; cd = tf; ci = ti;
          }
        }
      }
    }
    __syncthreads();
  }
#pragma unroll
  for (int i = 0; i < KNN; ++i) idx[((long)b * NN + n) * KNN + i] = bidx[i];
}

// ---------------------------------------------------------------------------
// pos_rel moments: stats[0..2]=S1, [3..8]=xx,yy,zz,xy,xz,yz
// ---------------------------------------------------------------------------
__global__ void pos_stats_kernel(const float* __restrict__ pos, const int* __restrict__ idx,
                                 float* __restrict__ stats) {
  float s[9] = {0, 0, 0, 0, 0, 0, 0, 0, 0};
  for (int g = blockIdx.x * blockDim.x + threadIdx.x; g < RROWS; g += gridDim.x * blockDim.x) {
    int b = g / (NN * KNN);
    int rem = g - b * (NN * KNN);
    int n = rem / KNN, kk = rem - n * KNN;
    int j = idx[((long)b * NN + n) * KNN + kk];
    float x = pos[((long)b * NN + n) * 3 + 0] - pos[((long)b * NN + j) * 3 + 0];
    float y = pos[((long)b * NN + n) * 3 + 1] - pos[((long)b * NN + j) * 3 + 1];
    float z = pos[((long)b * NN + n) * 3 + 2] - pos[((long)b * NN + j) * 3 + 2];
    s[0] += x; s[1] += y; s[2] += z;
    s[3] += x * x; s[4] += y * y; s[5] += z * z;
    s[6] += x * y; s[7] += x * z; s[8] += y * z;
  }
  __shared__ float red[9];
  if (threadIdx.x < 9) red[threadIdx.x] = 0.f;
  __syncthreads();
#pragma unroll
  for (int i = 0; i < 9; ++i) atomicAdd(&red[i], s[i]);
  __syncthreads();
  if (threadIdx.x < 9) atomicAdd(&stats[threadIdx.x], red[threadIdx.x]);
}

// Fold pos BN -> effective pos_w1/b1 (f32); pack pos_w2 fragment-linear bf16.
__global__ void fold_pos_bn(const float* __restrict__ stats, const float* __restrict__ w1,
                            const float* __restrict__ b1, const float* __restrict__ gam,
                            const float* __restrict__ bet, const float* __restrict__ w2,
                            float* __restrict__ w1e, float* __restrict__ b1e,
                            __bf16* __restrict__ w2p) {
  int c = threadIdx.x;  // 64 threads
  if (c < 64) {
    float invR = 1.f / (float)RROWS;
    float w0 = w1[0 * 64 + c], wy = w1[1 * 64 + c], wz = w1[2 * 64 + c];
    float m0 = stats[0] * invR, m1 = stats[1] * invR, m2 = stats[2] * invR;
    float mw = m0 * w0 + m1 * wy + m2 * wz;
    float Ey = mw + b1[c];
    float wSw = w0 * w0 * stats[3] + wy * wy * stats[4] + wz * wz * stats[5] +
                2.f * (w0 * wy * stats[6] + w0 * wz * stats[7] + wy * wz * stats[8]);
    float Ey2 = wSw * invR + 2.f * b1[c] * mw + b1[c] * b1[c];
    float var = fmaxf(Ey2 - Ey * Ey, 0.f);
    float s = gam[c] * rsqrtf(var + EPSV);
    w1e[0 * 64 + c] = w0 * s;
    w1e[1 * 64 + c] = wy * s;
    w1e[2 * 64 + c] = wz * s;
    b1e[c] = b1[c] * s + bet[c] - Ey * s;
    // pack pos_w2 [64,64] -> fragment-linear (KS=2)
    for (int p = c; p < 64 * 64; p += 64) {
      int e = p & 15, lane = (p >> 4) & 31, t2 = p >> 9;
      int ks = t2 & 1, nt = t2 >> 1;
      int k = ks * 32 + (lane >> 4) * 16 + e;
      int cc2 = nt * 16 + (lane & 15);
      w2p[p] = (__bf16)w2[k * 64 + cc2];
    }
  }
}

// ---------------------------------------------------------------------------
// pos_emb = relu(BN-folded(pos_rel@W1)) @ W2 + b2   -> posemb [R,64]
// grid R/64, block 128 (4 waves).
// ---------------------------------------------------------------------------
__global__ void posemb_kernel(const float* __restrict__ pos, const int* __restrict__ idx,
                              const float* __restrict__ w1e, const float* __restrict__ b1e,
                              const __bf16* __restrict__ w2p, const float* __restrict__ b2,
                              float* __restrict__ posemb) {
  __shared__ __bf16 pe_s[64][72];
  int g0 = blockIdx.x * 64;
  int rloc = threadIdx.x >> 1, half = threadIdx.x & 1;
  int g = g0 + rloc;
  int b = g / (NN * KNN);
  int rem = g - b * (NN * KNN);
  int n = rem / KNN, kk = rem - n * KNN;
  int j = idx[((long)b * NN + n) * KNN + kk];
  float px = pos[((long)b * NN + n) * 3 + 0] - pos[((long)b * NN + j) * 3 + 0];
  float py = pos[((long)b * NN + n) * 3 + 1] - pos[((long)b * NN + j) * 3 + 1];
  float pz = pos[((long)b * NN + n) * 3 + 2] - pos[((long)b * NN + j) * 3 + 2];
  for (int c = half * 32; c < half * 32 + 32; ++c) {
    float h = px * w1e[c] + py * w1e[64 + c] + pz * w1e[128 + c] + b1e[c];
    pe_s[rloc][c] = (__bf16)fmaxf(h, 0.f);
  }
  __syncthreads();
  int w = threadIdx.x >> 5, lane = threadIdx.x & 31;
  v8f acc[4] = {};
#pragma unroll
  for (int ks = 0; ks < 2; ++ks) {
    v16bf af = load_a_frag(&pe_s[w * 16][ks * 32], 72, lane);
#pragma unroll
    for (int nt = 0; nt < 4; ++nt) {
      v16bf bf_ = load_b_packed(w2p, nt * 2 + ks, lane);
      acc[nt] = wmma_bf16(af, bf_, acc[nt]);
    }
  }
  int cc = lane & 15, hh = lane >> 4;
#pragma unroll
  for (int nt = 0; nt < 4; ++nt)
#pragma unroll
    for (int v = 0; v < 8; ++v) {
      int row = g0 + w * 16 + v + 8 * hh;
      int col = nt * 16 + cc;
      posemb[(long)row * 64 + col] = acc[nt][v] + b2[col];
    }
}

// ---------------------------------------------------------------------------
// a_base = pos_emb + (q - k[idx]) + (up - up[idx]);  val = v[idx] + pos_emb + up_rel
// ---------------------------------------------------------------------------
__global__ void build_a_val(const float* __restrict__ posemb, const int* __restrict__ idx,
                            const float* __restrict__ qb, const float* __restrict__ kb,
                            const float* __restrict__ vb, const float* __restrict__ ub,
                            float* __restrict__ abase, float* __restrict__ valb) {
  for (long e = blockIdx.x * blockDim.x + threadIdx.x; e < (long)RROWS * 64;
       e += (long)gridDim.x * blockDim.x) {
    int g = (int)(e >> 6), c = (int)(e & 63);
    int b = g / (NN * KNN);
    int rem = g - b * (NN * KNN);
    int n = rem / KNN, kk = rem - n * KNN;
    long bn = (long)b * NN + n;
    int j = idx[bn * KNN + kk];
    long bj = (long)b * NN + j;
    float pe = posemb[e];
    float ur = ub[bn * 64 + c] - ub[bj * 64 + c];
    abase[e] = pe + (qb[bn * 64 + c] - kb[bj * 64 + c]) + ur;
    valb[e] = vb[bj * 64 + c] + pe + ur;
  }
}

// ---------------------------------------------------------------------------
// attn BN stats via WMMA: M2 = A^T A (64x64) and S1 = column sums over R rows.
// ---------------------------------------------------------------------------
__global__ void attn_stats(const float* __restrict__ abase, float* __restrict__ M2,
                           float* __restrict__ S1) {
  __shared__ __bf16 As[32][72];
  __shared__ float csum[64];
  if (threadIdx.x < 64) csum[threadIdx.x] = 0.f;
  __syncthreads();
  int w = threadIdx.x >> 5, lane = threadIdx.x & 31;
  v8f acc[4] = {};
  const int CHUNKS = RROWS / 32;
  for (int ch = blockIdx.x; ch < CHUNKS; ch += gridDim.x) {
    long g0 = (long)ch * 32;
    for (int i = threadIdx.x; i < 32 * 64; i += 128) {
      int r = i >> 6, c = i & 63;
      float x = abase[(g0 + r) * 64 + c];
      As[r][c] = (__bf16)x;
      atomicAdd(&csum[c], x);
    }
    __syncthreads();
    int r = lane & 15, h = lane >> 4;
    v16bf at;  // A^T fragment: rows = channels, K = chunk rows
#pragma unroll
    for (int v = 0; v < 8; ++v) {
      int k0 = ((v < 4) ? 0 : 16) + h * 8 + (v & 3) * 2;
      at[2 * v]     = As[k0][w * 16 + r];
      at[2 * v + 1] = As[k0 + 1][w * 16 + r];
    }
#pragma unroll
    for (int nt = 0; nt < 4; ++nt) {
      int c = lane & 15;
      v16bf bf_;
#pragma unroll
      for (int v = 0; v < 8; ++v) {
        int k = ((v < 4) ? 0 : 16) + h * 8 + (v & 3) * 2;
        bf_[2 * v]     = As[k][nt * 16 + c];
        bf_[2 * v + 1] = As[k + 1][nt * 16 + c];
      }
      acc[nt] = wmma_bf16(at, bf_, acc[nt]);
    }
    __syncthreads();
  }
  int cc = lane & 15, hh = lane >> 4;
#pragma unroll
  for (int nt = 0; nt < 4; ++nt)
#pragma unroll
    for (int v = 0; v < 8; ++v)
      atomicAdd(&M2[(w * 16 + v + 8 * hh) * 64 + nt * 16 + cc], acc[nt][v]);
  __syncthreads();
  if (threadIdx.x < 64) atomicAdd(&S1[threadIdx.x], csum[threadIdx.x]);
}

// Fold attn BN -> fragment-packed bf16 W1_eff / b1_eff; pack convT weights.
__global__ void fold_attn_bn(const float* __restrict__ w1, const float* __restrict__ b1,
                             const float* __restrict__ gam, const float* __restrict__ bet,
                             const float* __restrict__ M2, const float* __restrict__ S1,
                             const float* __restrict__ convt_w, __bf16* __restrict__ w1p,
                             float* __restrict__ b1eff, __bf16* __restrict__ cwp) {
  __shared__ float sM2[64 * 64];
  __shared__ float sS1[64];
  __shared__ float s_sh[256];
  for (int i = threadIdx.x; i < 4096; i += 256) sM2[i] = M2[i];
  if (threadIdx.x < 64) sS1[threadIdx.x] = S1[threadIdx.x];
  __syncthreads();
  int c = threadIdx.x;  // 256 hidden channels
  float invR = 1.f / (float)RROWS;
  float wv[64];
  for (int d = 0; d < 64; ++d) wv[d] = w1[d * 256 + c];
  float bb = b1[c];
  float mw = 0.f;
  for (int d = 0; d < 64; ++d) mw += sS1[d] * wv[d];
  mw *= invR;
  float sw = 0.f;
  for (int d = 0; d < 64; ++d) {
    float t = 0.f;
    for (int e = 0; e < 64; ++e) t += sM2[d * 64 + e] * wv[e];
    sw += wv[d] * t;
  }
  float Ey = mw + bb;
  float Ey2 = sw * invR + 2.f * bb * mw + bb * bb;
  float var = fmaxf(Ey2 - Ey * Ey, 0.f);
  float s = gam[c] * rsqrtf(var + EPSV);
  s_sh[c] = s;
  b1eff[c] = bb * s + bet[c] - Ey * s;
  __syncthreads();
  // W1_eff packed: Kd=64 (KS=2), Nd=256 (16 ntiles); tile = nt*2 + ks
  for (int p = threadIdx.x; p < 64 * 256; p += 256) {
    int e = p & 15, lane = (p >> 4) & 31, t2 = p >> 9;
    int ks = t2 & 1, nt = t2 >> 1;
    int k = ks * 32 + (lane >> 4) * 16 + e;
    int cc2 = nt * 16 + (lane & 15);
    w1p[p] = (__bf16)(w1[k * 256 + cc2] * s_sh[cc2]);
  }
  // convT packed: tile = (r*4 + nt)*8 + ks  (Kd=256 -> KS=8, Nd=64 -> 4 ntiles)
  for (int p = threadIdx.x; p < 2 * 256 * 64; p += 256) {
    int e = p & 15, lane = (p >> 4) & 31, t2 = p >> 9;
    int ks = t2 & 7;
    int ntr = t2 >> 3;
    int nt = ntr & 3, r = ntr >> 2;
    int k = ks * 32 + (lane >> 4) * 16 + e;
    int o = nt * 16 + (lane & 15);
    cwp[p] = (__bf16)convt_w[k * 128 + o * 2 + r];
  }
}

// ---------------------------------------------------------------------------
// Fused per-(b,n) attention tail. grid MROWS, block 128 (4 waves).
// ---------------------------------------------------------------------------
__global__ void fused_attn(const float* __restrict__ abase, const float* __restrict__ valb,
                           const __bf16* __restrict__ w1p, const float* __restrict__ b1eff,
                           const __bf16* __restrict__ cwp, const float* __restrict__ convt_b,
                           const float* __restrict__ end_w, const float* __restrict__ end_b,
                           const float* __restrict__ identity, float* __restrict__ out) {
  int bn = blockIdx.x;
  int b = bn >> 11, n = bn & (NN - 1);
  __shared__ __bf16 As[32][72];
  __shared__ __bf16 Hs[32][264];
  __shared__ float att[2][32][64];
  __shared__ float vals[KNN][64];
  __shared__ float agg[2][64];
  int t = threadIdx.x, w = t >> 5, lane = t & 31;
  long base = (long)bn * (KNN * 64);
  for (int i = t; i < 32 * 64; i += 128) {
    int r = i >> 6, c = i & 63;
    As[r][c] = (r < KNN) ? (__bf16)abase[base + (long)r * 64 + c] : (__bf16)0.f;
  }
  for (int i = t; i < KNN * 64; i += 128) vals[i >> 6][i & 63] = valb[base + i];
  __syncthreads();

  // GEMM1: H = relu(A @ W1eff + b1eff). 2 mtiles x 16 ntiles, K=64.
  v16bf af[2][2];
#pragma unroll
  for (int mt = 0; mt < 2; ++mt)
#pragma unroll
    for (int ks = 0; ks < 2; ++ks) af[mt][ks] = load_a_frag(&As[mt * 16][ks * 32], 72, lane);
#pragma unroll
  for (int i = 0; i < 8; ++i) {
    int tile = w + 4 * i;  // 0..31
    int mt = tile & 1, nt = tile >> 1;
    v8f acc = {};
#pragma unroll
    for (int ks = 0; ks < 2; ++ks)
      acc = wmma_bf16(af[mt][ks], load_b_packed(w1p, nt * 2 + ks, lane), acc);
    int cc = lane & 15, hh = lane >> 4;
#pragma unroll
    for (int v = 0; v < 8; ++v) {
      int row = mt * 16 + v + 8 * hh, col = nt * 16 + cc;
      Hs[row][col] = (__bf16)fmaxf(acc[v] + b1eff[col], 0.f);
    }
  }
  __syncthreads();

  // GEMM2: att[r] = H @ convW[r] + convt_b. 2r x 2mt x 4nt, K=256.
#pragma unroll
  for (int i = 0; i < 4; ++i) {
    int tile = w + 4 * i;  // 0..15
    int r = tile >> 3, mt = (tile >> 2) & 1, nt = tile & 3;
    v8f acc = {};
#pragma unroll
    for (int ks = 0; ks < 8; ++ks) {
      v16bf a2 = load_a_frag(&Hs[mt * 16][ks * 32], 264, lane);
      acc = wmma_bf16(a2, load_b_packed(cwp, (r * 4 + nt) * 8 + ks, lane), acc);
    }
    int cc = lane & 15, hh = lane >> 4;
#pragma unroll
    for (int v = 0; v < 8; ++v) {
      int row = mt * 16 + v + 8 * hh, col = nt * 16 + cc;
      att[r][row][col] = acc[v] + convt_b[col];
    }
  }
  __syncthreads();

  // softmax over K + aggregate: thread -> (r, o)
  {
    int r = t >> 6, o = t & 63;
    float m = -1e30f;
#pragma unroll
    for (int k2 = 0; k2 < KNN; ++k2) m = fmaxf(m, att[r][k2][o]);
    float e[KNN];
    float ssum = 0.f;
#pragma unroll
    for (int k2 = 0; k2 < KNN; ++k2) { e[k2] = __expf(att[r][k2][o] - m); ssum += e[k2]; }
    float inv = 1.f / ssum;
    float a = 0.f;
#pragma unroll
    for (int k2 = 0; k2 < KNN; ++k2) a += e[k2] * inv * vals[k2][o];
    agg[r][o] = a;
  }
  __syncthreads();

  // residual (64->128) + identity, thread -> out channel
  {
    int oc = t;  // 0..127
#pragma unroll
    for (int r = 0; r < 2; ++r) {
      float acc = end_b[oc];
      for (int o = 0; o < 64; ++o) acc += agg[r][o] * end_w[o * 128 + oc];
      long orow = (long)b * (NN * UPF) + n * 2 + r;
      out[orow * IND + oc] = identity[(long)bn * IND + oc] + acc;
    }
  }
}

// ---------------------------------------------------------------------------
extern "C" void kernel_launch(void* const* d_in, const int* in_sizes, int n_in,
                              void* d_out, int out_size, void* d_ws, size_t ws_size,
                              hipStream_t stream) {
  (void)in_sizes; (void)n_in; (void)out_size; (void)ws_size;
  const float* pos      = (const float*)d_in[0];
  const float* key_feat = (const float*)d_in[1];
  const float* query    = (const float*)d_in[2];
  const float* upfeat   = (const float*)d_in[3];
  const float* mv_w1 = (const float*)d_in[4];  const float* mv_b1 = (const float*)d_in[5];
  const float* mv_w2 = (const float*)d_in[6];  const float* mv_b2 = (const float*)d_in[7];
  const float* mv_ws = (const float*)d_in[8];  const float* mv_bs = (const float*)d_in[9];
  const float* q_w = (const float*)d_in[10];   const float* q_b = (const float*)d_in[11];
  const float* k_w = (const float*)d_in[12];   const float* k_b = (const float*)d_in[13];
  const float* v_w = (const float*)d_in[14];   const float* v_b = (const float*)d_in[15];
  const float* up_w = (const float*)d_in[16];  const float* up_b = (const float*)d_in[17];
  const float* pos_w1 = (const float*)d_in[18]; const float* pos_b1 = (const float*)d_in[19];
  const float* pos_g = (const float*)d_in[20];  const float* pos_be = (const float*)d_in[21];
  const float* pos_w2 = (const float*)d_in[22]; const float* pos_b2 = (const float*)d_in[23];
  const float* at_w1 = (const float*)d_in[24];  const float* at_b1 = (const float*)d_in[25];
  const float* at_g = (const float*)d_in[26];   const float* at_be = (const float*)d_in[27];
  const float* cw = (const float*)d_in[28];     const float* cb = (const float*)d_in[29];
  const float* end_w = (const float*)d_in[30];  const float* end_b = (const float*)d_in[31];
  float* out = (float*)d_out;

  char* ws = (char*)d_ws;
  size_t off = 0;
  auto alloc = [&](size_t bytes) { size_t o = off; off = (off + bytes + 255) & ~(size_t)255; return o; };
  float* kq     = (float*)(ws + alloc((size_t)MROWS * 256 * 4));
  float* hbuf   = (float*)(ws + alloc((size_t)MROWS * 128 * 4));
  float* value  = (float*)(ws + alloc((size_t)MROWS * 128 * 4));
  float* qbuf   = (float*)(ws + alloc((size_t)MROWS * 64 * 4));
  float* kbuf   = (float*)(ws + alloc((size_t)MROWS * 64 * 4));
  float* vbuf   = (float*)(ws + alloc((size_t)MROWS * 64 * 4));
  float* ubuf   = (float*)(ws + alloc((size_t)MROWS * 64 * 4));
  int*   idx    = (int*)  (ws + alloc((size_t)MROWS * KNN * 4));
  float* posemb = (float*)(ws + alloc((size_t)RROWS * 64 * 4));
  float* abase  = (float*)(ws + alloc((size_t)RROWS * 64 * 4));
  float* valb   = (float*)(ws + alloc((size_t)RROWS * 64 * 4));
  float* stats  = (float*)(ws + alloc(4224 * 4));   // [0..8]=pos, +16=S1(64), +80=M2(4096)
  float* S1     = stats + 16;
  float* M2     = stats + 80;
  // fragment-packed bf16 weights
  __bf16* mv_w1p = (__bf16*)(ws + alloc((size_t)256 * 128 * 2));
  __bf16* mv_wsp = (__bf16*)(ws + alloc((size_t)256 * 128 * 2));
  __bf16* mv_w2p = (__bf16*)(ws + alloc((size_t)128 * 128 * 2));
  __bf16* q_wp   = (__bf16*)(ws + alloc((size_t)128 * 64 * 2));
  __bf16* k_wp   = (__bf16*)(ws + alloc((size_t)128 * 64 * 2));
  __bf16* v_wp   = (__bf16*)(ws + alloc((size_t)128 * 64 * 2));
  __bf16* up_wp  = (__bf16*)(ws + alloc((size_t)128 * 64 * 2));
  __bf16* w1p    = (__bf16*)(ws + alloc((size_t)64 * 256 * 2));
  float*  b1eff  = (float*) (ws + alloc(256 * 4));
  __bf16* cwp    = (__bf16*)(ws + alloc((size_t)2 * 256 * 64 * 2));
  float*  posw1e = (float*) (ws + alloc(192 * 4));
  float*  posb1e = (float*) (ws + alloc(64 * 4));
  __bf16* posw2p = (__bf16*)(ws + alloc((size_t)64 * 64 * 2));

  hipMemsetAsync(stats, 0, 4224 * 4, stream);

  // Pack projection weights into fragment-linear bf16 (one-off, tiny)
  pack_weight<<<64, 256, 0, stream>>>(mv_w1, mv_w1p, 256, 128);
  pack_weight<<<64, 256, 0, stream>>>(mv_ws, mv_wsp, 256, 128);
  pack_weight<<<64, 256, 0, stream>>>(mv_w2, mv_w2p, 128, 128);
  pack_weight<<<32, 256, 0, stream>>>(q_w, q_wp, 128, 64);
  pack_weight<<<32, 256, 0, stream>>>(k_w, k_wp, 128, 64);
  pack_weight<<<32, 256, 0, stream>>>(v_w, v_wp, 128, 64);
  pack_weight<<<32, 256, 0, stream>>>(up_w, up_wp, 128, 64);

  // Point-wise projections (WMMA GEMMs, packed weights)
  concat_kernel<<<2048, 256, 0, stream>>>(key_feat, query, kq);
  wmma_gemm<true, false><<<dim3(MROWS / 64, 2), 128, 0, stream>>>(kq, mv_w1p, mv_b1, hbuf, MROWS, 256, 128);
  wmma_gemm<false, false><<<dim3(MROWS / 64, 2), 128, 0, stream>>>(kq, mv_wsp, mv_bs, value, MROWS, 256, 128);
  wmma_gemm<false, true><<<dim3(MROWS / 64, 2), 128, 0, stream>>>(hbuf, mv_w2p, mv_b2, value, MROWS, 128, 128);
  wmma_gemm<false, false><<<dim3(MROWS / 64, 1), 128, 0, stream>>>(query, q_wp, q_b, qbuf, MROWS, 128, 64);
  wmma_gemm<false, false><<<dim3(MROWS / 64, 1), 128, 0, stream>>>(key_feat, k_wp, k_b, kbuf, MROWS, 128, 64);
  wmma_gemm<false, false><<<dim3(MROWS / 64, 1), 128, 0, stream>>>(value, v_wp, v_b, vbuf, MROWS, 128, 64);
  wmma_gemm<false, false><<<dim3(MROWS / 64, 1), 128, 0, stream>>>(upfeat, up_wp, up_b, ubuf, MROWS, 128, 64);

  // KNN + position branch (BN folded via moment matrices)
  knn_kernel<<<BB * (NN / 128), 128, 0, stream>>>(pos, idx);
  pos_stats_kernel<<<256, 256, 0, stream>>>(pos, idx, stats);
  fold_pos_bn<<<1, 64, 0, stream>>>(stats, pos_w1, pos_b1, pos_g, pos_be, pos_w2,
                                    posw1e, posb1e, posw2p);
  posemb_kernel<<<RROWS / 64, 128, 0, stream>>>(pos, idx, posw1e, posb1e, posw2p, pos_b2, posemb);

  // Neighbor tensors + attn BN stats (A^T A via WMMA) + fold
  build_a_val<<<4096, 256, 0, stream>>>(posemb, idx, qbuf, kbuf, vbuf, ubuf, abase, valb);
  attn_stats<<<1024, 128, 0, stream>>>(abase, M2, S1);
  fold_attn_bn<<<1, 256, 0, stream>>>(at_w1, at_b1, at_g, at_be, M2, S1, cw, w1p, b1eff, cwp);

  // Fused attention tail
  fused_attn<<<MROWS, 128, 0, stream>>>(abase, valb, w1p, b1eff, cwp, cb, end_w, end_b,
                                        value, out);
}
